// RelationSAGELayer_14216341749897
// MI455X (gfx1250) — matrix operand order repaired and checked by reading, not compiled, not run on hardware
//
#include <hip/hip_runtime.h>

#define N_NODES 100000
#define N_EDGES 1600000
#define IN_DIM  128
#define OUT_DIM 128
#define REL_DIM 32
#define K_MSG   (IN_DIM + REL_DIM)   // 160
#define LDS_PITCH 168                // padded bf16 pitch: 336B rows, conflict-free

typedef __attribute__((ext_vector_type(16))) __bf16 v16bf;
typedef __attribute__((ext_vector_type(8)))  float  v8f;

__global__ void zero_ws_kernel(float* __restrict__ ws, long n) {
  long i = (long)blockIdx.x * blockDim.x + threadIdx.x;
  long stride = (long)gridDim.x * blockDim.x;
  for (; i < n; i += stride) ws[i] = 0.0f;
}

__global__ void degree_kernel(const int* __restrict__ edge_dst,
                              float* __restrict__ deg) {
  int i = blockIdx.x * blockDim.x + threadIdx.x;
  int stride = gridDim.x * blockDim.x;
  for (; i < N_EDGES; i += stride)
    atomicAdd(&deg[edge_dst[i]], 1.0f);
}

// One wave32 = one 16-edge tile: msg[16,128] = bf16-WMMA( [x[src]|rel_emb], msg_W^T )
// then scatter-add into agg[dst] via L2 float atomics.
__global__ __launch_bounds__(256)
void edge_msg_kernel(const float* __restrict__ x,
                     const int*   __restrict__ edge_src,
                     const int*   __restrict__ edge_dst,
                     const int*   __restrict__ rel_ids,
                     const float* __restrict__ rel_emb,
                     const float* __restrict__ msg_W,
                     const float* __restrict__ msg_b,
                     float*       __restrict__ agg) {
  __shared__ __bf16 Wl[OUT_DIM * LDS_PITCH];   // 42 KB of 320 KB WGP LDS
  // Stage msg_W ([128][160] fp32 row-major) into LDS as bf16, padded pitch.
  for (int i = threadIdx.x; i < OUT_DIM * K_MSG; i += blockDim.x) {
    int n = i / K_MSG, k = i - n * K_MSG;
    Wl[n * LDS_PITCH + k] = (__bf16)msg_W[i];
  }
  __syncthreads();

  const int lane  = threadIdx.x & 31;
  const int wave  = threadIdx.x >> 5;
  const int row   = lane & 15;   // A-matrix row owned by this lane
  const int khalf = lane >> 4;   // which K-half of the A layout
  const int wavesPerBlock = blockDim.x >> 5;
  const int nTiles = (N_EDGES + 15) / 16;

  for (int tile = blockIdx.x * wavesPerBlock + wave; tile < nTiles;
       tile += gridDim.x * wavesPerBlock) {
    const int ebase = tile * 16;
    int e = ebase + row; if (e >= N_EDGES) e = N_EDGES - 1;
    const int src = edge_src[e];
    const int rel = rel_ids[e];
    const float* xrow = x       + (size_t)src * IN_DIM;
    const float* rrow = rel_emb + (size_t)rel * REL_DIM;

    // A fragments: 5 K-steps of 16x32 bf16 (contiguous k order, matching B).
    // lane holds K = khalf*16 + g*8 + j (j=0..7) per group g, per k0.
    v16bf a[5];
    #pragma unroll
    for (int ks = 0; ks < 5; ++ks) {
      const int k0 = ks * 32;
      #pragma unroll
      for (int g = 0; g < 2; ++g) {
        const int kb = k0 + khalf * 16 + g * 8;
        const float* p = (kb < IN_DIM) ? (xrow + kb) : (rrow + (kb - IN_DIM));
        #pragma unroll
        for (int j = 0; j < 8; ++j) a[ks][g * 8 + j] = (__bf16)p[j];
      }
    }

    const int ncol = lane & 15;
    #pragma unroll
    for (int nt = 0; nt < 8; ++nt) {
      const int n = nt * 16 + ncol;
      v8f acc = {0.f, 0.f, 0.f, 0.f, 0.f, 0.f, 0.f, 0.f};
      #pragma unroll
      for (int ks = 0; ks < 5; ++ks) {
        // B 32x16 bf16: lane l holds N=l&15, K=(l>>4)*16 + j (16 contiguous)
        const int off = n * LDS_PITCH + ks * 32 + khalf * 16;
        v16bf b;
        #pragma unroll
        for (int j = 0; j < 16; ++j) b[j] = Wl[off + j];
        acc = __builtin_amdgcn_wmma_f32_16x16x32_bf16(
            false, a[ks], false, b, (short)0, acc, false, false);
      }
      const float bias = msg_b[n];
      // C layout: lane l, vgpr r -> M = r + 8*(l>>4), N = l&15
      #pragma unroll
      for (int r = 0; r < 8; ++r) {
        const int em = ebase + r + 8 * khalf;
        if (em < N_EDGES) {
          const int d = edge_dst[em];
          atomicAdd(&agg[(size_t)d * OUT_DIM + n], acc[r] + bias);
        }
      }
    }
  }
}

// One wave32 = one 16-node tile:
// out = relu( [x | agg/max(deg,1)] @ [self_W ; neigh_W]^T + self_b + neigh_b )
// Fused K=256 bf16 WMMA GEMM; B read from global (128 KB, L2/L0-hot).
__global__ __launch_bounds__(256)
void node_out_kernel(const float* __restrict__ x,
                     const float* __restrict__ agg,
                     const float* __restrict__ deg,
                     const float* __restrict__ self_W,
                     const float* __restrict__ self_b,
                     const float* __restrict__ neigh_W,
                     const float* __restrict__ neigh_b,
                     float*       __restrict__ out) {
  const int lane  = threadIdx.x & 31;
  const int wave  = threadIdx.x >> 5;
  const int row   = lane & 15;
  const int khalf = lane >> 4;
  const int wavesPerBlock = blockDim.x >> 5;
  const int nTiles = (N_NODES + 15) / 16;

  for (int tile = blockIdx.x * wavesPerBlock + wave; tile < nTiles;
       tile += gridDim.x * wavesPerBlock) {
    const int nbase = tile * 16;
    int node = nbase + row; if (node >= N_NODES) node = N_NODES - 1;
    const float invdeg = 1.0f / fmaxf(deg[node], 1.0f);
    const float* xrow = x   + (size_t)node * IN_DIM;
    const float* arow = agg + (size_t)node * OUT_DIM;

    // A = [x | agg*invdeg], K=256 -> 8 bf16 K-steps
    v16bf a[8];
    #pragma unroll
    for (int ks = 0; ks < 8; ++ks) {
      const int k0 = ks * 32;
      #pragma unroll
      for (int g = 0; g < 2; ++g) {
        const int kb = k0 + khalf * 16 + g * 8;
        if (kb < IN_DIM) {
          #pragma unroll
          for (int j = 0; j < 8; ++j)
            a[ks][g * 8 + j] = (__bf16)xrow[kb + j];
        } else {
          #pragma unroll
          for (int j = 0; j < 8; ++j)
            a[ks][g * 8 + j] = (__bf16)(arow[kb - IN_DIM + j] * invdeg);
        }
      }
    }

    const int ncol = lane & 15;
    #pragma unroll
    for (int nt = 0; nt < 8; ++nt) {
      const int n = nt * 16 + ncol;
      v8f acc = {0.f, 0.f, 0.f, 0.f, 0.f, 0.f, 0.f, 0.f};
      #pragma unroll
      for (int ks = 0; ks < 8; ++ks) {
        const int k0 = ks * 32;
        const float* Wp = (ks < 4)
            ? (self_W  + (size_t)n * IN_DIM  + (k0          + khalf * 16))
            : (neigh_W + (size_t)n * OUT_DIM + (k0 - IN_DIM + khalf * 16));
        v16bf b;
        #pragma unroll
        for (int j = 0; j < 16; ++j) b[j] = (__bf16)Wp[j];
        acc = __builtin_amdgcn_wmma_f32_16x16x32_bf16(
            false, a[ks], false, b, (short)0, acc, false, false);
      }
      const float bias = self_b[n] + neigh_b[n];
      #pragma unroll
      for (int r = 0; r < 8; ++r) {
        const int nm = nbase + r + 8 * khalf;
        if (nm < N_NODES)
          out[(size_t)nm * OUT_DIM + n] = fmaxf(acc[r] + bias, 0.0f);
      }
    }
  }
}

extern "C" void kernel_launch(void* const* d_in, const int* in_sizes, int n_in,
                              void* d_out, int out_size, void* d_ws, size_t ws_size,
                              hipStream_t stream) {
  (void)in_sizes; (void)n_in; (void)out_size; (void)ws_size;
  const float* x        = (const float*)d_in[0];
  const int*   edge_src = (const int*)  d_in[1];
  const int*   edge_dst = (const int*)  d_in[2];
  const int*   rel_ids  = (const int*)  d_in[3];
  const float* rel_emb  = (const float*)d_in[4];
  const float* msg_W    = (const float*)d_in[5];
  const float* msg_b    = (const float*)d_in[6];
  const float* self_W   = (const float*)d_in[7];
  const float* self_b   = (const float*)d_in[8];
  const float* neigh_W  = (const float*)d_in[9];
  const float* neigh_b  = (const float*)d_in[10];
  float* out = (float*)d_out;

  // Workspace layout: agg[N*128] | deg[N]  (~51.6 MB)
  float* agg = (float*)d_ws;
  float* deg = agg + (size_t)N_NODES * OUT_DIM;

  const long zcount = (long)N_NODES * OUT_DIM + N_NODES;
  zero_ws_kernel<<<2048, 256, 0, stream>>>(agg, zcount);
  degree_kernel<<<2048, 256, 0, stream>>>(edge_dst, deg);
  edge_msg_kernel<<<1536, 256, 0, stream>>>(x, edge_src, edge_dst, rel_ids,
                                            rel_emb, msg_W, msg_b, agg);
  node_out_kernel<<<800, 256, 0, stream>>>(x, agg, deg, self_W, self_b,
                                           neigh_W, neigh_b, out);
}
